// ProposalModule_76476187673126
// MI455X (gfx1250) — matrix operand order repaired and checked by reading, not compiled, not run on hardware
//
#include <hip/hip_runtime.h>
#include <hip/hip_bf16.h>

typedef __attribute__((ext_vector_type(16))) _Float16 v16h;
typedef __attribute__((ext_vector_type(8)))  float    v8f;
typedef __attribute__((ext_vector_type(4)))  unsigned u32x4;

#define B_        8
#define N_        32768
#define C_        256
#define P_        256
#define S_        16
#define OUTD      97
#define K0PAD     288          // 259 padded to multiple of 32
#define RADIUS_F  0.3f
#define BN_SCALE_F 0.9995003746879122f   // 1/sqrt(1+1e-3)
#define CHUNK     2048

// ---------------------------------------------------------------------------
// WMMA tile helpers (wave32, v_wmma_f32_16x16x32_f16)
// A (16xK f16) in LDS row-major [16][Kpad]; lane layout used here (contiguous K, not the ISA table order):
//   lanes 0-15: M=lane, VGPR j -> K pair (2j,2j+1)+16*hi (contiguous, matches B)
// B = W^T row-major [128][Kpad]; lanes 0-15 hold K=0..15, lanes 16-31 K=16..31
// ---------------------------------------------------------------------------
__device__ __forceinline__ v16h load_a16(const _Float16* As, int Kpad, int kb, int lane) {
  int m = lane & 15, hi = lane >> 4;
  union { v16h v; unsigned u[8]; } A;
  const _Float16* row = As + m * Kpad + kb + 16 * hi;
#pragma unroll
  for (int j = 0; j < 4; ++j) A.u[j] = *(const unsigned*)(row + 2 * j);
#pragma unroll
  for (int j = 4; j < 8; ++j) A.u[j] = *(const unsigned*)(row + 8 + 2 * (j - 4));
  return A.v;
}

__device__ __forceinline__ v16h load_b16(const _Float16* Wt, int Kpad, int n0, int kb, int lane) {
  int n = n0 + (lane & 15), hi = lane >> 4;
  union { v16h v; u32x4 q[2]; } Bv;
  const u32x4* p = (const u32x4*)(Wt + (size_t)n * Kpad + kb + 16 * hi);
  Bv.q[0] = p[0];
  Bv.q[1] = p[1];
  return Bv.v;
}

__device__ __forceinline__ v8f mm_tile(const _Float16* As, const _Float16* Wt,
                                       int Kpad, int n0, int lane) {
  v8f acc = {};
  for (int kb = 0; kb < Kpad; kb += 32) {
    v16h a = load_a16(As, Kpad, kb, lane);
    v16h b = load_b16(Wt, Kpad, n0, kb, lane);
    acc = __builtin_amdgcn_wmma_f32_16x16x32_f16(false, a, false, b, (short)0, acc,
                                                 false, false);
  }
  return acc;
}

// ---------------------------------------------------------------------------
// Prep: transpose weights to f16 W^T (row = out channel, contiguous K), pad;
// fold batch-norm into per-channel scale/shift.
// ---------------------------------------------------------------------------
__global__ void prep_kernel(
    const float* mw0, const float* mw1, const float* mw2,
    const float* hw1, const float* hw2, const float* hw3,
    const float* mb0, const float* mg0, const float* me0,
    const float* mb1, const float* mg1, const float* me1,
    const float* mb2, const float* mg2, const float* me2,
    const float* hb1, const float* hg1, const float* he1,
    const float* hb2, const float* hg2, const float* he2,
    _Float16* W0t, _Float16* W1t, _Float16* W2t,
    _Float16* H1t, _Float16* H2t, _Float16* H3t,
    float* sc, float* sh) {
  int i0 = blockIdx.x * blockDim.x + threadIdx.x;
  int stride = gridDim.x * blockDim.x;
  for (int j = i0; j < 128 * K0PAD; j += stride) {
    int n = j / K0PAD, k = j - n * K0PAD;
    W0t[j] = (k < 259) ? (_Float16)mw0[k * 128 + n] : (_Float16)0.f;
  }
  for (int j = i0; j < 128 * 128; j += stride) {
    int n = j >> 7, k = j & 127;
    W1t[j] = (_Float16)mw1[k * 128 + n];
    W2t[j] = (_Float16)mw2[k * 128 + n];
    H1t[j] = (_Float16)hw1[k * 128 + n];
    H2t[j] = (_Float16)hw2[k * 128 + n];
    H3t[j] = (n < OUTD) ? (_Float16)hw3[k * OUTD + n] : (_Float16)0.f;
  }
  const float* bs[5] = {mb0, mb1, mb2, hb1, hb2};
  const float* gs[5] = {mg0, mg1, mg2, hg1, hg2};
  const float* es[5] = {me0, me1, me2, he1, he2};
  for (int j = i0; j < 5 * 128; j += stride) {
    int L = j >> 7, n = j & 127;
    float g = gs[L][n];
    sc[j] = g * BN_SCALE_F;
    sh[j] = bs[L][n] * g * BN_SCALE_F + es[L][n];
  }
}

// ---------------------------------------------------------------------------
// Farthest point sampling: 1 workgroup (1024 thr = 32 waves) per batch.
// dist kept in registers (32 pts / lane), LDS tree argmax per step.
// ---------------------------------------------------------------------------
__global__ __launch_bounds__(1024) void fps_kernel(const float* __restrict__ xyz,
                                                   float* __restrict__ new_xyz) {
  const int b = blockIdx.x;
  const int tid = threadIdx.x;
  const float* X = xyz + (size_t)b * N_ * 3;
  float dist[32];
#pragma unroll
  for (int i = 0; i < 32; ++i) dist[i] = 1e10f;
  __shared__ float q[3];
  __shared__ float rv[1024];
  __shared__ int ri[1024];
  __shared__ int s_next;
  int last = 0;
  if (tid == 0) { q[0] = X[0]; q[1] = X[1]; q[2] = X[2]; }
  __syncthreads();
  for (int step = 0; step < P_; ++step) {
    if (tid == 0) {
      new_xyz[((size_t)b * P_ + step) * 3 + 0] = q[0];
      new_xyz[((size_t)b * P_ + step) * 3 + 1] = q[1];
      new_xyz[((size_t)b * P_ + step) * 3 + 2] = q[2];
    }
    float qa = q[0], qb = q[1], qc = q[2];
    float bestv = -1.f;
    int besti = 0;
#pragma unroll
    for (int i = 0; i < 32; ++i) {
      int p = tid + i * 1024;
      float dx = X[p * 3 + 0] - qa;
      float dy = X[p * 3 + 1] - qb;
      float dz = X[p * 3 + 2] - qc;
      float d = dx * dx + dy * dy + dz * dz;
      float nd = fminf(dist[i], d);
      dist[i] = nd;
      if (nd > bestv) { bestv = nd; besti = p; }
    }
    rv[tid] = bestv;
    ri[tid] = besti;
    __syncthreads();
    for (int off = 512; off > 0; off >>= 1) {
      if (tid < off) {
        float v2 = rv[tid + off];
        int i2 = ri[tid + off];
        if (v2 > rv[tid] || (v2 == rv[tid] && i2 < ri[tid])) { rv[tid] = v2; ri[tid] = i2; }
      }
      __syncthreads();
    }
    if (tid == 0) {
      int nx = ri[0];
      s_next = nx;
      q[0] = X[nx * 3 + 0];
      q[1] = X[nx * 3 + 1];
      q[2] = X[nx * 3 + 2];
    }
    __syncthreads();
    last = s_next;
    (void)last;
  }
}

// ---------------------------------------------------------------------------
// Ball query: 1 workgroup per batch, thread = query point. LDS-chunked xyz,
// ordered scan gives the first S in-range indices (== reference top_k(-key)).
// ---------------------------------------------------------------------------
__global__ __launch_bounds__(256) void ballq_kernel(const float* __restrict__ xyz,
                                                    const float* __restrict__ new_xyz,
                                                    int* __restrict__ idx) {
  const int b = blockIdx.x;
  const int p = threadIdx.x;
  const float* X = xyz + (size_t)b * N_ * 3;
  float qx = new_xyz[((size_t)b * P_ + p) * 3 + 0];
  float qy = new_xyz[((size_t)b * P_ + p) * 3 + 1];
  float qz = new_xyz[((size_t)b * P_ + p) * 3 + 2];
  const float r2 = RADIUS_F * RADIUS_F;
  int found = 0;
  int out[S_];
  __shared__ float chx[CHUNK], chy[CHUNK], chz[CHUNK];
  __shared__ int s_done;
  for (int base = 0; base < N_; base += CHUNK) {
    __syncthreads();
    if (p == 0) s_done = 0;
    __syncthreads();
    if (found >= S_) atomicAdd(&s_done, 1);
    for (int j = p; j < CHUNK; j += 256) {
      int gidx = base + j;
      chx[j] = X[gidx * 3 + 0];
      chy[j] = X[gidx * 3 + 1];
      chz[j] = X[gidx * 3 + 2];
    }
    __syncthreads();
    if (s_done == 256) break;  // uniform
    if (found < S_) {
      for (int j = 0; j < CHUNK && found < S_; ++j) {
        float dx = chx[j] - qx, dy = chy[j] - qy, dz = chz[j] - qz;
        if (dx * dx + dy * dy + dz * dz < r2) out[found++] = base + j;
      }
    }
  }
  int first = (found > 0) ? out[0] : 0;
  for (int s = 0; s < S_; ++s)
    idx[((size_t)b * P_ + p) * S_ + s] = (s < found) ? out[s] : first;
}

// ---------------------------------------------------------------------------
// Gather + 3-layer WMMA MLP + max-pool over S. 1 workgroup per (b,p) group,
// 8 waves x one 16x16 C-tile = 128 channels. M dim = S = 16 samples.
// ---------------------------------------------------------------------------
__global__ __launch_bounds__(256) void group_mlp_kernel(
    const float* __restrict__ xyz, const float* __restrict__ feat,
    const int* __restrict__ idx, const float* __restrict__ new_xyz,
    const _Float16* __restrict__ W0t, const _Float16* __restrict__ W1t,
    const _Float16* __restrict__ W2t, const float* __restrict__ sc,
    const float* __restrict__ sh, _Float16* __restrict__ pooled) {
  __shared__ _Float16 A0[S_ * K0PAD];
  __shared__ _Float16 A1[S_ * 128];
  __shared__ _Float16 A2[S_ * 128];
  __shared__ float pbuf[256];
  __shared__ int sidx[S_];
  __shared__ float sq[3];
  const int g = blockIdx.x;  // b*P + p
  const int b = g >> 8;
  const int tid = threadIdx.x;
  const int lane = tid & 31;
  const int w = tid >> 5;
  const int n0 = w * 16;
  if (tid < S_) sidx[tid] = idx[(size_t)g * S_ + tid];
  if (tid < 3) sq[tid] = new_xyz[(size_t)g * 3 + tid];
  __syncthreads();
  const float* Xb = xyz + (size_t)b * N_ * 3;
  const float* Fb = feat + (size_t)b * N_ * C_;
  for (int j = tid; j < S_ * K0PAD; j += 256) {
    int s = j / K0PAD, c = j - s * K0PAD;
    int pt = sidx[s];
    float v = 0.f;
    if (c < 3)        v = (Xb[pt * 3 + c] - sq[c]) * (1.0f / RADIUS_F);
    else if (c < 259) v = Fb[(size_t)pt * C_ + (c - 3)];
    A0[j] = (_Float16)v;
  }
  __syncthreads();
  const int col = n0 + (lane & 15);
  const int hi = lane >> 4;
  {  // layer 0: K=288
    union { v8f v; float f[8]; } acc;
    acc.v = mm_tile(A0, W0t, K0PAD, n0, lane);
    float s0 = sc[col], h0 = sh[col];
#pragma unroll
    for (int i = 0; i < 8; ++i) {
      float y = fmaxf(acc.f[i] * s0 + h0, 0.f);
      A1[(i + 8 * hi) * 128 + col] = (_Float16)y;
    }
  }
  __syncthreads();
  {  // layer 1: K=128
    union { v8f v; float f[8]; } acc;
    acc.v = mm_tile(A1, W1t, 128, n0, lane);
    float s1 = sc[128 + col], h1 = sh[128 + col];
#pragma unroll
    for (int i = 0; i < 8; ++i) {
      float y = fmaxf(acc.f[i] * s1 + h1, 0.f);
      A2[(i + 8 * hi) * 128 + col] = (_Float16)y;
    }
  }
  __syncthreads();
  {  // layer 2: K=128, fused max-pool over M (=S samples)
    union { v8f v; float f[8]; } acc;
    acc.v = mm_tile(A2, W2t, 128, n0, lane);
    float s2 = sc[256 + col], h2 = sh[256 + col];
    float pm = -1e30f;
#pragma unroll
    for (int i = 0; i < 8; ++i) {
      float y = fmaxf(acc.f[i] * s2 + h2, 0.f);
      pm = fmaxf(pm, y);
    }
    pbuf[tid] = pm;
  }
  __syncthreads();
  if (lane < 16) {
    float v = fmaxf(pbuf[w * 32 + lane], pbuf[w * 32 + lane + 16]);
    pooled[(size_t)g * 128 + n0 + lane] = (_Float16)v;
  }
}

// ---------------------------------------------------------------------------
// Head MLP on pooled features: 16 proposal rows per workgroup (M tile),
// 128->128->128->97(pad 128). Final layer: bias only, no BN/ReLU.
// ---------------------------------------------------------------------------
__global__ __launch_bounds__(256) void head_kernel(
    const _Float16* __restrict__ pooled, const _Float16* __restrict__ H1t,
    const _Float16* __restrict__ H2t, const _Float16* __restrict__ H3t,
    const float* __restrict__ sc, const float* __restrict__ sh,
    const float* __restrict__ hb3, float* __restrict__ net) {
  __shared__ _Float16 Ah[16 * 128];
  __shared__ _Float16 B1[16 * 128];
  __shared__ _Float16 B2[16 * 128];
  const int r0 = blockIdx.x * 16;
  const int tid = threadIdx.x;
  const int lane = tid & 31;
  const int w = tid >> 5;
  const int n0 = w * 16;
  for (int j = tid; j < 16 * 128; j += 256) Ah[j] = pooled[(size_t)r0 * 128 + j];
  __syncthreads();
  const int col = n0 + (lane & 15);
  const int hi = lane >> 4;
  {
    union { v8f v; float f[8]; } acc;
    acc.v = mm_tile(Ah, H1t, 128, n0, lane);
    float s3 = sc[384 + col], h3 = sh[384 + col];
#pragma unroll
    for (int i = 0; i < 8; ++i) {
      float y = fmaxf(acc.f[i] * s3 + h3, 0.f);
      B1[(i + 8 * hi) * 128 + col] = (_Float16)y;
    }
  }
  __syncthreads();
  {
    union { v8f v; float f[8]; } acc;
    acc.v = mm_tile(B1, H2t, 128, n0, lane);
    float s4 = sc[512 + col], h4 = sh[512 + col];
#pragma unroll
    for (int i = 0; i < 8; ++i) {
      float y = fmaxf(acc.f[i] * s4 + h4, 0.f);
      B2[(i + 8 * hi) * 128 + col] = (_Float16)y;
    }
  }
  __syncthreads();
  {
    union { v8f v; float f[8]; } acc;
    acc.v = mm_tile(B2, H3t, 128, n0, lane);
    if (col < OUTD) {
      float bv = hb3[col];
#pragma unroll
      for (int i = 0; i < 8; ++i)
        net[(size_t)(r0 + i + 8 * hi) * OUTD + col] = acc.f[i] + bv;
    }
  }
}

// ---------------------------------------------------------------------------
// Output scatter: tuple concatenated flat in return order.
// ---------------------------------------------------------------------------
__global__ void post_kernel(const float* __restrict__ net,
                            const float* __restrict__ new_xyz,
                            float* __restrict__ out) {
  int g = blockIdx.x * blockDim.x + threadIdx.x;
  if (g >= B_ * P_) return;
  const float* r = net + (size_t)g * OUTD;
  // obj (B,P,2) at 0
  out[g * 2 + 0] = r[0];
  out[g * 2 + 1] = r[1];
  // center (B,P,3) at 4096
  float* cen = out + 4096;
  for (int d = 0; d < 3; ++d) cen[g * 3 + d] = new_xyz[(size_t)g * 3 + d] + r[2 + d];
  // hs, hrn, hr (B,P,1)
  out[10240 + g] = r[5];
  out[12288 + g] = r[6];
  out[14336 + g] = r[6] * 3.14159265358979323846f;  // pi / NH, NH=1
  // ss (B,P,18) at 16384
  float* ss = out + 16384;
  for (int k = 0; k < 18; ++k) ss[g * 18 + k] = r[7 + k];
  // srn (B,P,18,3) at 53248 ; sr at 163840 (srn * MEAN_SIZE)
  float* srn = out + 53248;
  float* sr = out + 163840;
  for (int k = 0; k < 54; ++k) {
    float v = r[25 + k];
    srn[g * 54 + k] = v;
    sr[g * 54 + k] = v * ((float)(k % 5 + 1) * 0.2f);
  }
  // sem (B,P,18) at 274432
  float* sem = out + 274432;
  for (int k = 0; k < 18; ++k) sem[g * 18 + k] = r[79 + k];
}

// ---------------------------------------------------------------------------
extern "C" void kernel_launch(void* const* d_in, const int* in_sizes, int n_in,
                              void* d_out, int out_size, void* d_ws, size_t ws_size,
                              hipStream_t stream) {
  (void)in_sizes; (void)n_in; (void)out_size; (void)ws_size;
  const float* xyz  = (const float*)d_in[0];
  const float* feat = (const float*)d_in[1];
  const float* mw0 = (const float*)d_in[2],  *mb0 = (const float*)d_in[3];
  const float* mg0 = (const float*)d_in[4],  *me0 = (const float*)d_in[5];
  const float* mw1 = (const float*)d_in[6],  *mb1 = (const float*)d_in[7];
  const float* mg1 = (const float*)d_in[8],  *me1 = (const float*)d_in[9];
  const float* mw2 = (const float*)d_in[10], *mb2 = (const float*)d_in[11];
  const float* mg2 = (const float*)d_in[12], *me2 = (const float*)d_in[13];
  const float* hw1 = (const float*)d_in[14], *hb1 = (const float*)d_in[15];
  const float* hg1 = (const float*)d_in[16], *he1 = (const float*)d_in[17];
  const float* hw2 = (const float*)d_in[18], *hb2 = (const float*)d_in[19];
  const float* hg2 = (const float*)d_in[20], *he2 = (const float*)d_in[21];
  const float* hw3 = (const float*)d_in[22], *hb3 = (const float*)d_in[23];
  float* out = (float*)d_out;

  char* ws = (char*)d_ws;
  size_t off = 0;
  auto take = [&](size_t bytes) -> char* {
    char* p = ws + off;
    off = (off + bytes + 255) & ~(size_t)255;
    return p;
  };
  _Float16* W0t = (_Float16*)take(128 * K0PAD * sizeof(_Float16));
  _Float16* W1t = (_Float16*)take(128 * 128 * sizeof(_Float16));
  _Float16* W2t = (_Float16*)take(128 * 128 * sizeof(_Float16));
  _Float16* H1t = (_Float16*)take(128 * 128 * sizeof(_Float16));
  _Float16* H2t = (_Float16*)take(128 * 128 * sizeof(_Float16));
  _Float16* H3t = (_Float16*)take(128 * 128 * sizeof(_Float16));
  float* sc     = (float*)take(5 * 128 * sizeof(float));
  float* sh     = (float*)take(5 * 128 * sizeof(float));
  float* nxyz   = (float*)take((size_t)B_ * P_ * 3 * sizeof(float));
  int* idx      = (int*)take((size_t)B_ * P_ * S_ * sizeof(int));
  _Float16* pooled = (_Float16*)take((size_t)B_ * P_ * 128 * sizeof(_Float16));
  float* net    = (float*)take((size_t)B_ * P_ * OUTD * sizeof(float));

  prep_kernel<<<64, 256, 0, stream>>>(mw0, mw1, mw2, hw1, hw2, hw3,
                                      mb0, mg0, me0, mb1, mg1, me1, mb2, mg2, me2,
                                      hb1, hg1, he1, hb2, hg2, he2,
                                      W0t, W1t, W2t, H1t, H2t, H3t, sc, sh);
  fps_kernel<<<B_, 1024, 0, stream>>>(xyz, nxyz);
  ballq_kernel<<<B_, 256, 0, stream>>>(xyz, nxyz, idx);
  group_mlp_kernel<<<B_ * P_, 256, 0, stream>>>(xyz, feat, idx, nxyz,
                                                W0t, W1t, W2t, sc, sh, pooled);
  head_kernel<<<(B_ * P_) / 16, 256, 0, stream>>>(pooled, H1t, H2t, H3t, sc, sh,
                                                  hb3, net);
  post_kernel<<<(B_ * P_ + 255) / 256, 256, 0, stream>>>(net, nxyz, out);
}
